// FragAlignLoss_53231824667134
// MI455X (gfx1250) — compile-verified, matrix operand order checked by reading, not run on hardware
//
#include <hip/hip_runtime.h>

// Problem constants (from reference setup_inputs): B=256, F=32, N=32, D=256
#define BB    256
#define FF    32
#define NN    32
#define DD    256
#define MROWS (BB * FF)   // 8192 face rows
#define NROWS (BB * NN)   // 8192 ner rows

// Block tile: 128 x 128, K staged 64 at a time through LDS (bf16 hi/lo limbs).
#define TM    128
#define TN    128
#define TK    64
#define STR   72          // padded row stride in ushorts: 72*2=144B (16B aligned, bank-spread)

typedef float  v8f   __attribute__((ext_vector_type(8)));
typedef __bf16 v16bf __attribute__((ext_vector_type(16)));
typedef unsigned short u16x8 __attribute__((ext_vector_type(8)));
typedef unsigned short u16x4 __attribute__((ext_vector_type(4)));

union Frag { v16bf v; u16x8 h[2]; };

// Truncating fp32 -> bf16 split helpers (integer-only, no __bf16 scalar math).
__device__ __forceinline__ unsigned short bf_hi(float x) {
  union { float f; unsigned u; } c; c.f = x;
  return (unsigned short)(c.u >> 16);
}
__device__ __forceinline__ float bf_hif(float x) {
  union { float f; unsigned u; } c; c.f = x;
  c.u &= 0xFFFF0000u;
  return c.f;
}

// ---------------------------------------------------------------------------
// Kernel 1: zero the scalar output, compute valid[m] = (sum_d face[m,d] != 0)
// ---------------------------------------------------------------------------
__global__ void fragalign_prep(const float* __restrict__ face,
                               float* __restrict__ valid,
                               float* __restrict__ out) {
  int row = blockIdx.x * blockDim.x + threadIdx.x;   // 0..8191
  if (row == 0) out[0] = 0.0f;
  if (row < MROWS) {
    const float4* p = (const float4*)(face + (size_t)row * DD);
    float s = 0.0f;
#pragma unroll 8
    for (int j = 0; j < DD / 4; ++j) {
      float4 v = p[j];
      s += v.x + v.y + v.z + v.w;
    }
    valid[row] = (s != 0.0f) ? 1.0f : 0.0f;
  }
}

// ---------------------------------------------------------------------------
// Kernel 2: split-fp32 (bf16x2) GEMM on v_wmma_f32_16x16x32_bf16
//           + relu/==1/valid transform + full reduction
// ---------------------------------------------------------------------------
__global__ void __launch_bounds__(256, 2)
fragalign_main(const float* __restrict__ face,
               const float* __restrict__ ner,
               const float* __restrict__ valid,
               float* __restrict__ out) {
  __shared__ __align__(16) unsigned short Ahi[TM * STR];
  __shared__ __align__(16) unsigned short Alo[TM * STR];
  __shared__ __align__(16) unsigned short Bhi[TN * STR];
  __shared__ __align__(16) unsigned short Blo[TN * STR];
  __shared__ float validS[TM];
  __shared__ float red[256];

  const int t    = threadIdx.x;
  const int lane = t & 31;
  const int wave = t >> 5;            // 0..7
  const int wm   = wave & 3;          // 4 waves over M -> 32 rows each
  const int wn   = wave >> 2;         // 2 waves over N -> 64 cols each
  const int ll   = lane & 15;         // row/col within 16 (A: M, B: N)
  const int lh   = lane >> 4;         // K-run selector (0/1)

  const int blockM = blockIdx.x * TM;
  const int blockN = blockIdx.y * TN;

  if (t < TM) validS[t] = valid[blockM + t];

  v8f acc[2][4];
#pragma unroll
  for (int mi = 0; mi < 2; ++mi)
#pragma unroll
    for (int ni = 0; ni < 4; ++ni)
      acc[mi][ni] = (v8f){0.f, 0.f, 0.f, 0.f, 0.f, 0.f, 0.f, 0.f};

  // global->LDS staging: 128 rows x 64 cols per matrix, as float4 + bf16 split
  const int c4 = t & 15;              // float4 column group 0..15
  const int r0 = t >> 4;              // base row 0..15

  for (int kb = 0; kb < DD; kb += TK) {   // 4 K-stages
    __syncthreads();
#pragma unroll
    for (int i = 0; i < 8; ++i) {
      const int row = r0 + i * 16;
      float4 av = *(const float4*)(face + (size_t)(blockM + row) * DD + kb + c4 * 4);
      float4 bv = *(const float4*)(ner  + (size_t)(blockN + row) * DD + kb + c4 * 4);
      u16x4 ah, al, bh, bl;
      ah[0]=bf_hi(av.x); al[0]=bf_hi(av.x-bf_hif(av.x));
      ah[1]=bf_hi(av.y); al[1]=bf_hi(av.y-bf_hif(av.y));
      ah[2]=bf_hi(av.z); al[2]=bf_hi(av.z-bf_hif(av.z));
      ah[3]=bf_hi(av.w); al[3]=bf_hi(av.w-bf_hif(av.w));
      bh[0]=bf_hi(bv.x); bl[0]=bf_hi(bv.x-bf_hif(bv.x));
      bh[1]=bf_hi(bv.y); bl[1]=bf_hi(bv.y-bf_hif(bv.y));
      bh[2]=bf_hi(bv.z); bl[2]=bf_hi(bv.z-bf_hif(bv.z));
      bh[3]=bf_hi(bv.w); bl[3]=bf_hi(bv.w-bf_hif(bv.w));
      const int o = row * STR + c4 * 4;
      *(u16x4*)&Ahi[o] = ah;  *(u16x4*)&Alo[o] = al;
      *(u16x4*)&Bhi[o] = bh;  *(u16x4*)&Blo[o] = bl;
      if (kb + TK < DD) {   // prefetch next K stage (global_prefetch_b8)
        __builtin_prefetch(face + (size_t)(blockM + row) * DD + kb + TK + c4 * 4, 0, 1);
        __builtin_prefetch(ner  + (size_t)(blockN + row) * DD + kb + TK + c4 * 4, 0, 1);
      }
    }
    __syncthreads();

#pragma unroll
    for (int k32 = 0; k32 < TK; k32 += 32) {
      const int base = k32 + lh * 8;  // lane's first 8-element K-run (second at +16)
      Frag ahi[2], alo[2];
#pragma unroll
      for (int mi = 0; mi < 2; ++mi) {
        const int o = (wm * 32 + mi * 16 + ll) * STR + base;
        ahi[mi].h[0] = *(const u16x8*)&Ahi[o];
        ahi[mi].h[1] = *(const u16x8*)&Ahi[o + 16];
        alo[mi].h[0] = *(const u16x8*)&Alo[o];
        alo[mi].h[1] = *(const u16x8*)&Alo[o + 16];
      }
#pragma unroll
      for (int ni = 0; ni < 4; ++ni) {
        const int o = (wn * 64 + ni * 16 + ll) * STR + base;
        Frag bhi, blo;
        bhi.h[0] = *(const u16x8*)&Bhi[o];
        bhi.h[1] = *(const u16x8*)&Bhi[o + 16];
        blo.h[0] = *(const u16x8*)&Blo[o];
        blo.h[1] = *(const u16x8*)&Blo[o + 16];
#pragma unroll
        for (int mi = 0; mi < 2; ++mi) {
          // dot = hiA*hiB + hiA*loB + loA*hiB  (error ~2^-16 relative)
          acc[mi][ni] = __builtin_amdgcn_wmma_f32_16x16x32_bf16(
              false, ahi[mi].v, false, bhi.v, (short)0, acc[mi][ni], false, false);
          acc[mi][ni] = __builtin_amdgcn_wmma_f32_16x16x32_bf16(
              false, ahi[mi].v, false, blo.v, (short)0, acc[mi][ni], false, false);
          acc[mi][ni] = __builtin_amdgcn_wmma_f32_16x16x32_bf16(
              false, alo[mi].v, false, bhi.v, (short)0, acc[mi][ni], false, false);
        }
      }
    }
  }

  // Epilogue: s = relu(1 -/+ dot); s==1 -> 0; weight by valid[m]; local sum.
  float lsum = 0.0f;
#pragma unroll
  for (int mi = 0; mi < 2; ++mi) {
#pragma unroll
    for (int ni = 0; ni < 4; ++ni) {
#pragma unroll
      for (int i = 0; i < 8; ++i) {
        const int rlocal = wm * 32 + mi * 16 + i + lh * 8;   // C/D: M = i + 8*(lane>=16)
        const int gm = blockM + rlocal;
        const int gn = blockN + wn * 64 + ni * 16 + ll;      // N = lane%16
        const int bface = gm >> 5;                            // F = 32
        const int bner  = gn >> 5;                            // N = 32
        const float dot = acc[mi][ni][i];
        float s = (bface == bner) ? (1.0f - dot) : (1.0f + dot);
        s = fmaxf(s, 0.0f);
        s = (s == 1.0f) ? 0.0f : s;
        lsum += s * validS[rlocal];
      }
    }
  }

  red[t] = lsum;
  __syncthreads();
#pragma unroll
  for (int s = 128; s > 0; s >>= 1) {
    if (t < s) red[t] += red[t + s];
    __syncthreads();
  }
  if (t == 0) atomicAdd(out, red[0] * (1.0f / (float)BB));
}

// ---------------------------------------------------------------------------
extern "C" void kernel_launch(void* const* d_in, const int* in_sizes, int n_in,
                              void* d_out, int out_size, void* d_ws, size_t ws_size,
                              hipStream_t stream) {
  (void)in_sizes; (void)n_in; (void)out_size; (void)ws_size;
  const float* face = (const float*)d_in[0];   // [B,F,D] fp32
  const float* ner  = (const float*)d_in[1];   // [B,N,D] fp32
  float* out   = (float*)d_out;                // scalar fp32
  float* valid = (float*)d_ws;                 // 8192 floats scratch

  fragalign_prep<<<MROWS / 256, 256, 0, stream>>>(face, valid, out);

  dim3 grid(MROWS / TM, NROWS / TN);           // 64 x 64 blocks
  fragalign_main<<<grid, 256, 0, stream>>>(face, ner, valid, out);
}
